// ScaledDotProductAttention_6262062317851
// MI455X (gfx1250) — matrix operand order repaired and checked by reading, not compiled, not run on hardware
//
#include <hip/hip_runtime.h>

typedef __attribute__((ext_vector_type(16))) __bf16          v16bf;
typedef __attribute__((ext_vector_type(16))) unsigned short  v16us;
typedef __attribute__((ext_vector_type(8)))  float           v8f;

#define BB 16
#define LL 2048
#define DD 1024
#define NT 512          // 16 waves: fills all 4 SIMDs of the WGP 4-deep,
                        // since 192KB LDS allows only 1 workgroup per WGP
#define NW (NT / 32)
#define SCALE 0.03125f  // 1/sqrt(1024)

__device__ __forceinline__ unsigned short bf16u(float f) {
  __bf16 h = (__bf16)f;
  return __builtin_bit_cast(unsigned short, h);
}

// Load a 16-bit fragment (16 bf16) from two 16-byte-aligned LDS chunks.
__device__ __forceinline__ v16bf load_frag16(const unsigned short* c0,
                                             const unsigned short* c1) {
  union { v16bf v; uint4 q[2]; } u;
  u.q[0] = *(const uint4*)c0;
  u.q[1] = *(const uint4*)c1;
  return u.v;
}

// Convert 16 f32 -> v16bf by element; clang fuses pairs into v_cvt_pk_bf16_f32.
__device__ __forceinline__ v16bf cvt_frag16(const float* f) {
  v16bf r;
#pragma unroll
  for (int i = 0; i < 16; ++i) r[i] = (__bf16)f[i];
  return r;
}

__global__ __launch_bounds__(NT)
void sdpa_bf16_wmma_kernel(const float* __restrict__ q,
                           const float* __restrict__ k,
                           const float* __restrict__ v,
                           const int*   __restrict__ mask,
                           float* __restrict__ out,    // [B,L,D]
                           float* __restrict__ attn)   // [B,L,L]
{
  __shared__ float sc[16][LL];                 // 128 KB score/prob stripe (f32)
  __shared__ union {
    unsigned short qbf[16][DD];                // 32 KB: bf16 Q tile (phase 1)
    unsigned short pbf[16][LL];                // 64 KB: bf16 P tile (phase 3)
  } sh;                                        // total LDS: 192 KB -> 1 WG/WGP

  const int b    = blockIdx.x >> 7;            // 128 row-tiles per batch
  const int row0 = (blockIdx.x & 127) << 4;
  const int w    = threadIdx.x >> 5;           // wave id 0..15
  const int lane = threadIdx.x & 31;
  const int nn   = lane & 15;
  const int hi   = lane >> 4;
  const int akoff = hi * 16;                   // A-fragment K sub-offset per half-wave

  const float* qb = q + (size_t)b * LL * DD;
  const float* kb = k + (size_t)b * LL * DD;
  const float* vb = v + (size_t)b * LL * DD;

  // ---------------- Stage Q tile as bf16 into LDS (contiguous, coalesced) ----------------
  {
    const float4* src = (const float4*)(qb + (size_t)row0 * DD);
    uint4* dst = (uint4*)sh.qbf;               // 8 bf16 per uint4
    for (int i = threadIdx.x; i < (16 * DD) / 8; i += NT) {
      float4 f0 = src[2 * i], f1 = src[2 * i + 1];
      uint4 o;
      o.x = ((unsigned)bf16u(f0.y) << 16) | bf16u(f0.x);
      o.y = ((unsigned)bf16u(f0.w) << 16) | bf16u(f0.z);
      o.z = ((unsigned)bf16u(f1.y) << 16) | bf16u(f1.x);
      o.w = ((unsigned)bf16u(f1.w) << 16) | bf16u(f1.z);
      dst[i] = o;
    }
  }
  __syncthreads();

  // ---------------- Phase 1: S = (Q K^T) * scale, masked -> LDS ----------------
  {
    int mk = mask[b * LL + (w << 4) + nn];     // first tile's mask, prefetched
    for (int jt = w; jt < LL / 16; jt += NW) {
      const int    col0 = jt << 4;
      const float* krow = kb + (size_t)(col0 + nn) * DD + hi * 16;
      // software-pipeline: issue next tile's mask load + K prefetch now,
      // let 32 WMMAs hide the latency.
      int mk_next = 0;
      if (jt + NW < LL / 16) {
        mk_next = mask[b * LL + ((jt + NW) << 4) + nn];
        __builtin_prefetch(kb + (size_t)(col0 + 16 * NW + nn) * DD + hi * 16, 0, 1);
      }

      v8f acc = {};
      for (int ks = 0; ks < DD; ks += 32) {
        v16bf av = load_frag16(&sh.qbf[nn][ks + akoff],
                               &sh.qbf[nn][ks + 8 + akoff]);
        float bfv[16];
#pragma unroll
        for (int i = 0; i < 4; ++i)
          *(float4*)(bfv + 4 * i) = ((const float4*)(krow + ks))[i];
        v16bf bv = cvt_frag16(bfv);

        acc = __builtin_amdgcn_wmma_f32_16x16x32_bf16(
            false, av, false, bv, (short)0, acc, false, false);
      }
      const int col = col0 + nn;
#pragma unroll
      for (int r = 0; r < 8; ++r) {
        float val = mk ? acc[r] * SCALE : -__builtin_inff();
        sc[r + 8 * hi][col] = val;
      }
      mk = mk_next;
    }
  }
  __syncthreads();

  // ---------------- Phase 2: row softmax (float4); write attn_weight + bf16 P ----------------
  for (int r = w; r < 16; r += NW) {           // one row per wave
    float mx = -__builtin_inff();
    for (int c = lane * 4; c < LL; c += 128) {
      float4 f = *(const float4*)&sc[r][c];
      mx = fmaxf(mx, fmaxf(fmaxf(f.x, f.y), fmaxf(f.z, f.w)));
    }
#pragma unroll
    for (int off = 16; off; off >>= 1) mx = fmaxf(mx, __shfl_xor(mx, off, 32));

    float s = 0.f;
    for (int c = lane * 4; c < LL; c += 128) {
      float4 f = *(const float4*)&sc[r][c];
      f.x = __expf(f.x - mx); f.y = __expf(f.y - mx);
      f.z = __expf(f.z - mx); f.w = __expf(f.w - mx);
      *(float4*)&sc[r][c] = f;
      s += (f.x + f.y) + (f.z + f.w);
    }
#pragma unroll
    for (int off = 16; off; off >>= 1) s += __shfl_xor(s, off, 32);

    const float inv  = 1.0f / s;
    float*      arow = attn + ((size_t)b * LL + row0 + r) * LL;
    for (int c = lane * 4; c < LL; c += 128) {
      float4 f = *(const float4*)&sc[r][c];
      f.x *= inv; f.y *= inv; f.z *= inv; f.w *= inv;
      *(float4*)(arow + c) = f;                // 512B/wave attn_weight store
      uint2 p2;
      p2.x = ((unsigned)bf16u(f.y) << 16) | bf16u(f.x);
      p2.y = ((unsigned)bf16u(f.w) << 16) | bf16u(f.z);
      *(uint2*)&sh.pbf[r][c] = p2;             // bf16 P for phase 3
    }
  }
  __syncthreads();

  // ---------------- Phase 3: O = P V ----------------
  {
    const int kb16 = hi * 16;
    for (int nt = w; nt < DD / 16; nt += NW) {
      const int ncol0 = nt << 4;
      v8f acc = {};
      for (int ks = 0; ks < LL; ks += 32) {
        v16bf av = load_frag16(&sh.pbf[nn][ks + akoff],
                               &sh.pbf[nn][ks + 8 + akoff]);
        float bfv[16];
#pragma unroll
        for (int i = 0; i < 16; ++i)
          bfv[i] = vb[(size_t)(ks + kb16 + i) * DD + ncol0 + nn];
        v16bf bv = cvt_frag16(bfv);

        acc = __builtin_amdgcn_wmma_f32_16x16x32_bf16(
            false, av, false, bv, (short)0, acc, false, false);
      }
      float* orow = out + ((size_t)b * LL + row0) * DD;
#pragma unroll
      for (int r = 0; r < 8; ++r)
        orow[(size_t)(r + 8 * hi) * DD + ncol0 + nn] = acc[r];
    }
  }
}

extern "C" void kernel_launch(void* const* d_in, const int* in_sizes, int n_in,
                              void* d_out, int out_size, void* d_ws, size_t ws_size,
                              hipStream_t stream) {
  (void)in_sizes; (void)n_in; (void)d_ws; (void)ws_size; (void)out_size;
  const float* q    = (const float*)d_in[0];
  const float* k    = (const float*)d_in[1];
  const float* v    = (const float*)d_in[2];
  const int*   mask = (const int*)d_in[3];
  float* out  = (float*)d_out;                       // [B,L,D] first
  float* attn = out + (size_t)BB * LL * DD;          // then [B,L,L]

  dim3 grid(BB * (LL / 16));                         // 2048 workgroups, 16 waves each
  sdpa_bf16_wmma_kernel<<<grid, NT, 0, stream>>>(q, k, v, mask, out, attn);
}
